// LigRecConv_73031623901835
// MI455X (gfx1250) — matrix operand order repaired and checked by reading, not compiled, not run on hardware
//
#include <hip/hip_runtime.h>
#include <hip/hip_bf16.h>

typedef __attribute__((ext_vector_type(16))) _Float16 v16h;
typedef __attribute__((ext_vector_type(8)))  _Float16 v8h;
typedef __attribute__((ext_vector_type(4)))  _Float16 v4h;
typedef __attribute__((ext_vector_type(8)))  float    v8f;

#define N_LIG_C 10000
#define N_REC_C 40000
#define E_LL_C  262144
#define E_RL_C  262144

__device__ __forceinline__ float silu_f(float x) {
    return x / (1.0f + __expf(-x));
}

__device__ __forceinline__ v16h cat16(v8h lo, v8h hi) {
    return __builtin_shufflevector(lo, hi, 0,1,2,3,4,5,6,7,8,9,10,11,12,13,14,15);
}

// A fragment (16x32 f16) from an LDS row-major tile.
// ISA layout: lanes 0-15 hold row M=lane, halves = K {0..7,16..23}; lanes 16-31
// hold row M=lane-16, halves = K {8..15,24..31} (per 32-wide K tile).
__device__ __forceinline__ v16h load_a_lds(const _Float16* base, int strideH, int kt, int lane) {
    const int row = lane & 15;
    const int k0  = kt * 32 + ((lane & 16) ? 8 : 0);
    const _Float16* p = base + row * strideH + k0;
    v8h lo = *(const v8h*)(p);
    v8h hi = *(const v8h*)(p + 16);
    return cat16(lo, hi);
}

// B fragment (32x16 f16) from pre-packed global weights: per (kt,nt,lane) 16
// contiguous halves = column N = nt*16 + lane%16, K = kt*32 + 8*(lane/16) + j%8 + 16*(j/8).
__device__ __forceinline__ v16h load_b_pk(const _Float16* pw, int kt, int nt, int lane) {
    const _Float16* p = pw + ((((kt << 3) + nt) << 5) + lane) * 16;
    v8h lo = *(const v8h*)(p);
    v8h hi = *(const v8h*)(p + 8);
    return cat16(lo, hi);
}

__device__ __forceinline__ v8f wmma32(v16h a, v16h b, v8f c) {
    return __builtin_amdgcn_wmma_f32_16x16x32_f16(false, a, false, b, (short)0, c, false, false);
}

__device__ __forceinline__ void atomic_add_f32(float* p, float v) {
    __hip_atomic_fetch_add(p, v, __ATOMIC_RELAXED, __HIP_MEMORY_SCOPE_AGENT);
}

// ---------------------------------------------------------------------------
// Pack fp32 [K x 128] weight (row-major, K = input dim) into f16 B-fragment
// order, zero-padding K up to KT*32.
// ---------------------------------------------------------------------------
__global__ void lrc_pack_w(const float* __restrict__ src, _Float16* __restrict__ dst,
                           int K, int KT) {
    int idx = blockIdx.x * blockDim.x + threadIdx.x;
    int total = KT * 4096;                    // KT * 8 ntiles * 32 lanes * 16 halves
    if (idx >= total) return;
    int j    = idx & 15;
    int lane = (idx >> 4) & 31;
    int nt   = (idx >> 9) & 7;
    int kt   = idx >> 12;
    int k = kt * 32 + ((lane & 16) ? 8 : 0) + (j & 7) + ((j & 8) << 1);
    int n = nt * 16 + (lane & 15);
    float v = (k < K) ? src[k * 128 + n] : 0.0f;
    dst[idx] = (_Float16)v;
}

__global__ void lrc_zero(float* __restrict__ p, int n) {
    int i = blockIdx.x * blockDim.x + threadIdx.x;
    if (i < n) p[i] = 0.0f;
}

// ---------------------------------------------------------------------------
// Edge kernel: one wave32 per 16-edge tile. Computes both the edge MLP
// (msg_h, scattered by f32 atomics into h_neigh) and the coord MLP
// (coef -> msg_x, scattered into x_neigh).
// ---------------------------------------------------------------------------
__global__ __launch_bounds__(32) void lrc_edge_kernel(
    const float* __restrict__ h_src_tab, const float* __restrict__ h_dst_tab,
    const float* __restrict__ x_src_tab, const float* __restrict__ x_dst_tab,
    const int* __restrict__ src, const int* __restrict__ dst,
    const _Float16* __restrict__ pw1e, const float* __restrict__ b1e,
    const _Float16* __restrict__ pw2e, const float* __restrict__ b2e,
    const _Float16* __restrict__ pw1c, const float* __restrict__ b1c,
    const _Float16* __restrict__ pw2c, const float* __restrict__ b2c,
    const float* __restrict__ w3,
    float* __restrict__ h_neigh, float* __restrict__ x_neigh)
{
    const int AS = 296;   // f-tile row stride in halves (592B: 16B-aligned, bank-staggered)
    const int CS = 136;   // act-tile row stride in halves (272B)
    __shared__ alignas(16) _Float16 Af[16 * AS];   // f = [h_src | h_dst | dij | 0pad] 16x288
    __shared__ alignas(16) _Float16 act[16 * CS];  // hidden activations 16x128
    __shared__ int   s_dst[16];
    __shared__ int   s_src[16];
    __shared__ float s_xd[16 * 3];

    const int lane = threadIdx.x;
    const int e0   = blockIdx.x * 16;

    // --- stage geometry + dij + feature padding ---
    if (lane < 16) {
        int e = e0 + lane;
        int s = src[e], d = dst[e];
        s_src[lane] = s; s_dst[lane] = d;
        float dx = x_src_tab[s * 3 + 0] - x_dst_tab[d * 3 + 0];
        float dy = x_src_tab[s * 3 + 1] - x_dst_tab[d * 3 + 1];
        float dz = x_src_tab[s * 3 + 2] - x_dst_tab[d * 3 + 2];
        float dij = sqrtf(dx * dx + dy * dy + dz * dz);
        float inv = 1.0f / (dij + 1e-9f);
        s_xd[lane * 3 + 0] = dx * inv;
        s_xd[lane * 3 + 1] = dy * inv;
        s_xd[lane * 3 + 2] = dz * inv;
        Af[lane * AS + 256] = (_Float16)dij;
        #pragma unroll
        for (int j = 257; j < 288; j++) Af[lane * AS + j] = (_Float16)0.0f;
    }
    __syncthreads();

    // --- gather endpoint features (2 lanes per edge, 64 floats each half) ---
    {
        const int el = lane >> 1, hf = lane & 1;
        const int s = s_src[el], d = s_dst[el];
        const float4* ps = (const float4*)(h_src_tab + s * 128 + hf * 64);
        const float4* pd = (const float4*)(h_dst_tab + d * 128 + hf * 64);
        _Float16* rowA = Af + el * AS + hf * 64;
        #pragma unroll
        for (int j = 0; j < 16; j++) {
            float4 v = ps[j];
            v4h h4 = { (_Float16)v.x, (_Float16)v.y, (_Float16)v.z, (_Float16)v.w };
            *(v4h*)(rowA + j * 4) = h4;
            float4 w = pd[j];
            v4h g4 = { (_Float16)w.x, (_Float16)w.y, (_Float16)w.z, (_Float16)w.w };
            *(v4h*)(rowA + 128 + j * 4) = g4;
        }
    }
    __syncthreads();

    const int colLo = lane & 15;
    const int mHi   = (lane & 16) ? 8 : 0;   // C/D row offset for this half-wave

    // ===================== edge MLP =====================
    // layer1: [16x288] @ [288x128] -> SiLU -> act
    #pragma unroll 1
    for (int nt = 0; nt < 8; nt++) {
        v8f acc = {0.f,0.f,0.f,0.f,0.f,0.f,0.f,0.f};
        #pragma unroll
        for (int kt = 0; kt < 9; kt++)
            acc = wmma32(load_a_lds(Af, AS, kt, lane), load_b_pk(pw1e, kt, nt, lane), acc);
        float bias = b1e[nt * 16 + colLo];
        #pragma unroll
        for (int r = 0; r < 8; r++)
            act[(r + mHi) * CS + nt * 16 + colLo] = (_Float16)silu_f(acc[r] + bias);
    }
    __syncthreads();
    // layer2: [16x128] @ [128x128] -> SiLU -> atomic scatter (segment_sum)
    #pragma unroll 1
    for (int nt = 0; nt < 8; nt++) {
        v8f acc = {0.f,0.f,0.f,0.f,0.f,0.f,0.f,0.f};
        #pragma unroll
        for (int kt = 0; kt < 4; kt++)
            acc = wmma32(load_a_lds(act, CS, kt, lane), load_b_pk(pw2e, kt, nt, lane), acc);
        float bias = b2e[nt * 16 + colLo];
        int col = nt * 16 + colLo;
        #pragma unroll
        for (int r = 0; r < 8; r++) {
            float m = silu_f(acc[r] + bias);
            atomic_add_f32(&h_neigh[s_dst[r + mHi] * 128 + col], m);
        }
    }
    __syncthreads();

    // ===================== coord MLP =====================
    #pragma unroll 1
    for (int nt = 0; nt < 8; nt++) {
        v8f acc = {0.f,0.f,0.f,0.f,0.f,0.f,0.f,0.f};
        #pragma unroll
        for (int kt = 0; kt < 9; kt++)
            acc = wmma32(load_a_lds(Af, AS, kt, lane), load_b_pk(pw1c, kt, nt, lane), acc);
        float bias = b1c[nt * 16 + colLo];
        #pragma unroll
        for (int r = 0; r < 8; r++)
            act[(r + mHi) * CS + nt * 16 + colLo] = (_Float16)silu_f(acc[r] + bias);
    }
    __syncthreads();
    float cacc[8] = {0.f,0.f,0.f,0.f,0.f,0.f,0.f,0.f};
    #pragma unroll 1
    for (int nt = 0; nt < 8; nt++) {
        v8f acc = {0.f,0.f,0.f,0.f,0.f,0.f,0.f,0.f};
        #pragma unroll
        for (int kt = 0; kt < 4; kt++)
            acc = wmma32(load_a_lds(act, CS, kt, lane), load_b_pk(pw2c, kt, nt, lane), acc);
        float bias = b2c[nt * 16 + colLo];
        float w3v = w3[nt * 16 + colLo];
        #pragma unroll
        for (int r = 0; r < 8; r++)
            cacc[r] += silu_f(acc[r] + bias) * w3v;   // fused @ w3 (128 -> 1)
    }
    // reduce coef across the 16 lanes of each half-wave
    #pragma unroll
    for (int r = 0; r < 8; r++) {
        float v = cacc[r];
        v += __shfl_xor(v, 8, 16);
        v += __shfl_xor(v, 4, 16);
        v += __shfl_xor(v, 2, 16);
        v += __shfl_xor(v, 1, 16);
        cacc[r] = v;
    }
    if (colLo == 0) {   // lanes 0 and 16 hold edges 0..7 / 8..15
        #pragma unroll
        for (int r = 0; r < 8; r++) {
            int e = r + mHi;
            int d = s_dst[e];
            float c = cacc[r];
            atomic_add_f32(&x_neigh[d * 3 + 0], c * s_xd[e * 3 + 0]);
            atomic_add_f32(&x_neigh[d * 3 + 1], c * s_xd[e * 3 + 1]);
            atomic_add_f32(&x_neigh[d * 3 + 2], c * s_xd[e * 3 + 2]);
        }
    }
}

// ---------------------------------------------------------------------------
// Node update: [h_lig | h_neigh] (16x256) -> MLP -> residual add; coords add.
// ---------------------------------------------------------------------------
__global__ __launch_bounds__(32) void lrc_node_kernel(
    const float* __restrict__ h_lig, const float* __restrict__ x_lig,
    const float* __restrict__ h_neigh, const float* __restrict__ x_neigh,
    const _Float16* __restrict__ pw1, const float* __restrict__ b1,
    const _Float16* __restrict__ pw2, const float* __restrict__ b2,
    float* __restrict__ out_h, float* __restrict__ out_x)
{
    const int AS = 264;
    const int CS = 136;
    __shared__ alignas(16) _Float16 Af[16 * AS];
    __shared__ alignas(16) _Float16 act[16 * CS];

    const int lane = threadIdx.x;
    const int n0 = blockIdx.x * 16;

    {
        const int el = lane >> 1, hf = lane & 1;
        const int node = n0 + el;
        const float4* ph = (const float4*)(h_lig   + node * 128 + hf * 64);
        const float4* pn = (const float4*)(h_neigh + node * 128 + hf * 64);
        _Float16* rowA = Af + el * AS + hf * 64;
        #pragma unroll
        for (int j = 0; j < 16; j++) {
            float4 v = ph[j];
            v4h h4 = { (_Float16)v.x, (_Float16)v.y, (_Float16)v.z, (_Float16)v.w };
            *(v4h*)(rowA + j * 4) = h4;
            float4 w = pn[j];
            v4h g4 = { (_Float16)w.x, (_Float16)w.y, (_Float16)w.z, (_Float16)w.w };
            *(v4h*)(rowA + 128 + j * 4) = g4;
        }
    }
    __syncthreads();

    const int colLo = lane & 15;
    const int mHi   = (lane & 16) ? 8 : 0;

    #pragma unroll 1
    for (int nt = 0; nt < 8; nt++) {
        v8f acc = {0.f,0.f,0.f,0.f,0.f,0.f,0.f,0.f};
        #pragma unroll
        for (int kt = 0; kt < 8; kt++)
            acc = wmma32(load_a_lds(Af, AS, kt, lane), load_b_pk(pw1, kt, nt, lane), acc);
        float bias = b1[nt * 16 + colLo];
        #pragma unroll
        for (int r = 0; r < 8; r++)
            act[(r + mHi) * CS + nt * 16 + colLo] = (_Float16)silu_f(acc[r] + bias);
    }
    __syncthreads();
    #pragma unroll 1
    for (int nt = 0; nt < 8; nt++) {
        v8f acc = {0.f,0.f,0.f,0.f,0.f,0.f,0.f,0.f};
        #pragma unroll
        for (int kt = 0; kt < 4; kt++)
            acc = wmma32(load_a_lds(act, CS, kt, lane), load_b_pk(pw2, kt, nt, lane), acc);
        float bias = b2[nt * 16 + colLo];
        int col = nt * 16 + colLo;
        #pragma unroll
        for (int r = 0; r < 8; r++) {
            int node = n0 + r + mHi;
            out_h[node * 128 + col] = h_lig[node * 128 + col] + acc[r] + bias;
        }
    }
    if (lane < 16) {
        int node = n0 + lane;
        out_x[node * 3 + 0] = x_lig[node * 3 + 0] + x_neigh[node * 3 + 0];
        out_x[node * 3 + 1] = x_lig[node * 3 + 1] + x_neigh[node * 3 + 1];
        out_x[node * 3 + 2] = x_lig[node * 3 + 2] + x_neigh[node * 3 + 2];
    }
}

// ---------------------------------------------------------------------------
extern "C" void kernel_launch(void* const* d_in, const int* in_sizes, int n_in,
                              void* d_out, int out_size, void* d_ws, size_t ws_size,
                              hipStream_t stream) {
    const float* h_lig = (const float*)d_in[0];
    const float* h_rec = (const float*)d_in[1];
    const float* x_lig = (const float*)d_in[2];
    const float* x_rec = (const float*)d_in[3];
    const int* ll_src = (const int*)d_in[4];
    const int* ll_dst = (const int*)d_in[5];
    const int* rl_src = (const int*)d_in[6];
    const int* rl_dst = (const int*)d_in[7];
    // params in setup_inputs() insertion order:
    const float* ell_w1 = (const float*)d_in[8];
    const float* ell_b1 = (const float*)d_in[9];
    const float* ell_w2 = (const float*)d_in[10];
    const float* ell_b2 = (const float*)d_in[11];
    const float* cll_w1 = (const float*)d_in[12];
    const float* cll_b1 = (const float*)d_in[13];
    const float* cll_w2 = (const float*)d_in[14];
    const float* cll_b2 = (const float*)d_in[15];
    const float* cll_w3 = (const float*)d_in[16];
    const float* erl_w1 = (const float*)d_in[17];
    const float* erl_b1 = (const float*)d_in[18];
    const float* erl_w2 = (const float*)d_in[19];
    const float* erl_b2 = (const float*)d_in[20];
    const float* crl_w1 = (const float*)d_in[21];
    const float* crl_b1 = (const float*)d_in[22];
    const float* crl_w2 = (const float*)d_in[23];
    const float* crl_b2 = (const float*)d_in[24];
    const float* crl_w3 = (const float*)d_in[25];
    const float* nod_w1 = (const float*)d_in[26];
    const float* nod_b1 = (const float*)d_in[27];
    const float* nod_w2 = (const float*)d_in[28];
    const float* nod_b2 = (const float*)d_in[29];

    char* W = (char*)d_ws;
    _Float16* pw1_ell = (_Float16*)(W + 0);        // 9*4096 halves = 73728 B each
    _Float16* pw1_cll = (_Float16*)(W + 73728);
    _Float16* pw1_erl = (_Float16*)(W + 147456);
    _Float16* pw1_crl = (_Float16*)(W + 221184);
    _Float16* pw1_nod = (_Float16*)(W + 294912);   // 8*4096 halves = 65536 B
    _Float16* pw2_ell = (_Float16*)(W + 360448);   // 4*4096 halves = 32768 B each
    _Float16* pw2_cll = (_Float16*)(W + 393216);
    _Float16* pw2_erl = (_Float16*)(W + 425984);
    _Float16* pw2_crl = (_Float16*)(W + 458752);
    _Float16* pw2_nod = (_Float16*)(W + 491520);
    float* h_neigh = (float*)(W + 524288);               // 10000*128 f32
    float* x_neigh = (float*)(W + 524288 + 5120000);     // 10000*3 f32 (contiguous)

    float* out = (float*)d_out;
    float* out_hlig = out;                     // [10000,128]
    float* out_hrec = out + 1280000;           // [40000,128]
    float* out_xlig = out + 6400000;           // [10000,3]
    float* out_xrec = out + 6430000;           // [40000,3]

    auto cdiv = [](int a, int b) { return (a + b - 1) / b; };

    // pack all WMMA B operands (fp32 -> f16 fragment order)
    lrc_pack_w<<<cdiv(9 * 4096, 256), 256, 0, stream>>>(ell_w1, pw1_ell, 257, 9);
    lrc_pack_w<<<cdiv(9 * 4096, 256), 256, 0, stream>>>(cll_w1, pw1_cll, 257, 9);
    lrc_pack_w<<<cdiv(9 * 4096, 256), 256, 0, stream>>>(erl_w1, pw1_erl, 257, 9);
    lrc_pack_w<<<cdiv(9 * 4096, 256), 256, 0, stream>>>(crl_w1, pw1_crl, 257, 9);
    lrc_pack_w<<<cdiv(8 * 4096, 256), 256, 0, stream>>>(nod_w1, pw1_nod, 256, 8);
    lrc_pack_w<<<cdiv(4 * 4096, 256), 256, 0, stream>>>(ell_w2, pw2_ell, 128, 4);
    lrc_pack_w<<<cdiv(4 * 4096, 256), 256, 0, stream>>>(cll_w2, pw2_cll, 128, 4);
    lrc_pack_w<<<cdiv(4 * 4096, 256), 256, 0, stream>>>(erl_w2, pw2_erl, 128, 4);
    lrc_pack_w<<<cdiv(4 * 4096, 256), 256, 0, stream>>>(crl_w2, pw2_crl, 128, 4);
    lrc_pack_w<<<cdiv(4 * 4096, 256), 256, 0, stream>>>(nod_w2, pw2_nod, 128, 4);

    // zero accumulators (h_neigh + x_neigh contiguous)
    lrc_zero<<<cdiv(N_LIG_C * 128 + N_LIG_C * 3, 256), 256, 0, stream>>>(
        h_neigh, N_LIG_C * 128 + N_LIG_C * 3);

    // pass-through outputs
    hipMemcpyAsync(out_hrec, h_rec, (size_t)N_REC_C * 128 * sizeof(float),
                   hipMemcpyDeviceToDevice, stream);
    hipMemcpyAsync(out_xrec, x_rec, (size_t)N_REC_C * 3 * sizeof(float),
                   hipMemcpyDeviceToDevice, stream);

    // edge message passing (lig->lig, rec->lig)
    lrc_edge_kernel<<<E_LL_C / 16, 32, 0, stream>>>(
        h_lig, h_lig, x_lig, x_lig, ll_src, ll_dst,
        pw1_ell, ell_b1, pw2_ell, ell_b2,
        pw1_cll, cll_b1, pw2_cll, cll_b2, cll_w3,
        h_neigh, x_neigh);
    lrc_edge_kernel<<<E_RL_C / 16, 32, 0, stream>>>(
        h_rec, h_lig, x_rec, x_lig, rl_src, rl_dst,
        pw1_erl, erl_b1, pw2_erl, erl_b2,
        pw1_crl, crl_b1, pw2_crl, crl_b2, crl_w3,
        h_neigh, x_neigh);

    // node update + residuals
    lrc_node_kernel<<<N_LIG_C / 16, 32, 0, stream>>>(
        h_lig, x_lig, h_neigh, x_neigh,
        pw1_nod, nod_b1, pw2_nod, nod_b2,
        out_hlig, out_xlig);
}
